// HGTLayer_61907658605250
// MI455X (gfx1250) — compile-verified, matrix operand order checked by reading, not run on hardware
//
#include <hip/hip_runtime.h>
#include <hip/hip_bf16.h>
#include <math.h>

typedef __attribute__((ext_vector_type(16))) __bf16 v16bf;
typedef __attribute__((ext_vector_type(8)))  float  v8f;

#define NN 50000
#define DD 256
#define HH 8
#define DKK 32
#define EE 500000

// ---------------------------------------------------------------------------
// float atomic max via int/uint monotone-bits trick (init with -inf)
// ---------------------------------------------------------------------------
__device__ __forceinline__ void atomic_max_f32(float* addr, float val) {
  if (val >= 0.0f) atomicMax((int*)addr, __float_as_int(val));
  else             atomicMin((unsigned int*)addr, (unsigned int)__float_as_int(val));
}

// ---------------------------------------------------------------------------
// A-fragment loader: 16x32 bf16 fragment from row-major f32 (ld = 256).
// Wave32 layout (ISA 7.12.2): lanes 0-15 row M=lane, K = {kb..kb+7, kb+16..kb+23}
// with kb = 0 for lanes 0-15 and kb = 8 for lanes 16-31 (caller pre-offsets).
// ---------------------------------------------------------------------------
__device__ __forceinline__ v16bf load_afrag(const float* __restrict__ ap) {
  float4 a0 = *(const float4*)(ap);
  float4 a1 = *(const float4*)(ap + 4);
  float4 a2 = *(const float4*)(ap + 16);
  float4 a3 = *(const float4*)(ap + 20);
  v16bf f;
  f[0]  = (__bf16)a0.x; f[1]  = (__bf16)a0.y; f[2]  = (__bf16)a0.z; f[3]  = (__bf16)a0.w;
  f[4]  = (__bf16)a1.x; f[5]  = (__bf16)a1.y; f[6]  = (__bf16)a1.z; f[7]  = (__bf16)a1.w;
  f[8]  = (__bf16)a2.x; f[9]  = (__bf16)a2.y; f[10] = (__bf16)a2.z; f[11] = (__bf16)a2.w;
  f[12] = (__bf16)a3.x; f[13] = (__bf16)a3.y; f[14] = (__bf16)a3.z; f[15] = (__bf16)a3.w;
  return f;
}

// ---------------------------------------------------------------------------
// Fuse per-head transforms into the projection weights:
//   WkF[et][i][h*32+j] = sum_m Wk[et][i][h*32+m] * w_att[et][h][m][j]
//   WvF analogously with w_msg; biases fused the same way.
// (source node type st == et in this model)
// ---------------------------------------------------------------------------
__global__ __launch_bounds__(256) void fuse_weights_kernel(
    const float* __restrict__ Wk, const float* __restrict__ bk,
    const float* __restrict__ Wv, const float* __restrict__ bv,
    const float* __restrict__ watt, const float* __restrict__ wmsg,
    float* __restrict__ WkF, float* __restrict__ bkF,
    float* __restrict__ WvF, float* __restrict__ bvF) {
  int idx = blockIdx.x * 256 + threadIdx.x;           // 2*256*256 total
  if (idx >= 2 * DD * DD) return;
  int et  = idx >> 16;
  int rem = idx & 0xFFFF;
  int i   = rem >> 8;
  int c   = rem & 255;
  int h   = c >> 5;
  int j   = c & 31;
  const float* wa = watt + (size_t)((et * HH + h) * DKK) * DKK + j; // [m][j], stride 32
  const float* wm = wmsg + (size_t)((et * HH + h) * DKK) * DKK + j;
  const float* wkr = Wk + ((size_t)et * DD + i) * DD + h * DKK;
  const float* wvr = Wv + ((size_t)et * DD + i) * DD + h * DKK;
  float ak = 0.f, av = 0.f;
#pragma unroll 4
  for (int m = 0; m < DKK; ++m) {
    ak = fmaf(wkr[m], wa[m * DKK], ak);
    av = fmaf(wvr[m], wm[m * DKK], av);
  }
  WkF[idx] = ak;
  WvF[idx] = av;
  if (i == 0) {
    const float* bkp = bk + et * DD + h * DKK;
    const float* bvp = bv + et * DD + h * DKK;
    float sk = 0.f, sv = 0.f;
    for (int m = 0; m < DKK; ++m) {
      sk = fmaf(bkp[m], wa[m * DKK], sk);
      sv = fmaf(bvp[m], wm[m * DKK], sv);
    }
    bkF[et * DD + c] = sk;
    bvF[et * DD + c] = sv;
  }
}

// ---------------------------------------------------------------------------
// Pre-pack a 256x256 f32 B matrix into bf16 wave32 WMMA B-fragments:
// out[((tile*8 + kblk)*32 + lane)*16 + e], 16 col-tiles x 8 k-blocks.
// B-fragment mirrors A: lane holds col n = tile*16 + (l&15),
// K = {k0+kb .. +7, k0+16+kb .. +7}, kb = 8*(l>=16).
// ---------------------------------------------------------------------------
__global__ void pack_b_kernel(const float* __restrict__ B, __bf16* __restrict__ out) {
  int t = blockIdx.x;                 // 0..127 : tile = t>>3, kblk = t&7
  int l = threadIdx.x;                // 0..31
  int tile = t >> 3, kblk = t & 7;
  int n  = tile * 16 + (l & 15);
  int kb = (l >> 4) * 8;
  int k0 = kblk * 32;
  __bf16* o = out + ((size_t)t * 32 + l) * 16;
#pragma unroll
  for (int i = 0; i < 8; ++i) {
    o[i]     = (__bf16)B[(size_t)(k0 + kb + i) * DD + n];
    o[8 + i] = (__bf16)B[(size_t)(k0 + 16 + kb + i) * DD + n];
  }
}

// ---------------------------------------------------------------------------
// C[N x 256] = A[N x 256] * B + bias  (bf16 WMMA, f32 accumulate)
// Block: 16 rows x 256 cols, 8 waves, each wave owns 2 16x16 col tiles.
// ---------------------------------------------------------------------------
__global__ __launch_bounds__(256) void wmma_gemm_bias_kernel(
    const float* __restrict__ A, const __bf16* __restrict__ Bp,
    const float* __restrict__ bias, float* __restrict__ C) {
  const int w    = threadIdx.x >> 5;
  const int l    = threadIdx.x & 31;
  const int lrow = l & 15;
  const int hi   = l >> 4;
  const int m0   = blockIdx.x * 16;
  v8f c0 = {0, 0, 0, 0, 0, 0, 0, 0};
  v8f c1 = {0, 0, 0, 0, 0, 0, 0, 0};
  const float*  arow = A + (size_t)(m0 + lrow) * DD + hi * 8;
  const __bf16* bp0  = Bp + ((size_t)(2 * w) * 8 * 32 + l) * 16;
  const __bf16* bp1  = Bp + ((size_t)(2 * w + 1) * 8 * 32 + l) * 16;
#pragma unroll
  for (int kk = 0; kk < 8; ++kk) {
    v16bf af = load_afrag(arow + kk * 32);
    v16bf b0 = *(const v16bf*)(bp0 + (size_t)kk * 32 * 16);
    v16bf b1 = *(const v16bf*)(bp1 + (size_t)kk * 32 * 16);
    c0 = __builtin_amdgcn_wmma_f32_16x16x32_bf16(false, af, false, b0, (short)0, c0, false, false);
    c1 = __builtin_amdgcn_wmma_f32_16x16x32_bf16(false, af, false, b1, (short)0, c1, false, false);
  }
  const int   n0 = w * 32 + lrow, n1 = n0 + 16;
  const float bb0 = bias[n0], bb1 = bias[n1];
  const int   rbase = m0 + hi * 8;
#pragma unroll
  for (int r = 0; r < 8; ++r) {
    C[(size_t)(rbase + r) * DD + n0] = c0[r] + bb0;
    C[(size_t)(rbase + r) * DD + n1] = c1[r] + bb1;
  }
}

// ---------------------------------------------------------------------------
// Per-etype init: softmax max = -inf, sum = 0, destination accumulator = 0.
// grid = NN*DD/256 exactly.
// ---------------------------------------------------------------------------
__global__ void init_et_kernel(float* __restrict__ mbuf, float* __restrict__ sbuf,
                               float* __restrict__ hout_dt) {
  int i = blockIdx.x * 256 + threadIdx.x;
  if (i < NN * HH) { mbuf[i] = -__builtin_inff(); sbuf[i] = 0.0f; }
  hout_dt[i] = 0.0f;
}

// ---------------------------------------------------------------------------
// Pass A: one wave per edge. Full 1KB row gathers of Q[dst], K[src];
// per-head dot via shfl reduction over 4-lane groups; atomic max per (dst,h).
// ---------------------------------------------------------------------------
__global__ __launch_bounds__(256) void edge_scores_kernel(
    const float* __restrict__ Q, const float* __restrict__ K,
    const int* __restrict__ src, const int* __restrict__ dst,
    const float* __restrict__ mu, float mscale,
    float* __restrict__ scores, float* __restrict__ mbuf) {
  int wid = blockIdx.x * 8 + (threadIdx.x >> 5);
  if (wid >= EE) return;
  int l  = threadIdx.x & 31;
  int sN = src[wid], dN = dst[wid];
  const float4* qr = (const float4*)(Q + (size_t)dN * DD);
  const float4* kr = (const float4*)(K + (size_t)sN * DD);
  float4 q0 = qr[2 * l], q1 = qr[2 * l + 1];
  float4 k0 = kr[2 * l], k1 = kr[2 * l + 1];
  float p = q0.x * k0.x + q0.y * k0.y + q0.z * k0.z + q0.w * k0.w +
            q1.x * k1.x + q1.y * k1.y + q1.z * k1.z + q1.w * k1.w;
  p += __shfl_xor(p, 1, 32);
  p += __shfl_xor(p, 2, 32);
  if ((l & 3) == 0) {
    int   h = l >> 2;
    float t = p * mu[h] * mscale;
    scores[(size_t)wid * HH + h] = t;
    atomic_max_f32(mbuf + (size_t)dN * HH + h, t);
  }
}

// ---------------------------------------------------------------------------
// Pass B: e = exp(score - max[dst,h]); accumulate segment sums.
// ---------------------------------------------------------------------------
__global__ __launch_bounds__(256) void edge_exp_kernel(
    float* __restrict__ scores, const int* __restrict__ dst,
    const float* __restrict__ mbuf, float* __restrict__ sbuf) {
  int i = blockIdx.x * 256 + threadIdx.x;
  if (i >= EE * HH) return;
  int e = i >> 3, h = i & 7;
  int dN = dst[e];
  float ex = __expf(scores[i] - mbuf[(size_t)dN * HH + h]);
  scores[i] = ex;
  atomicAdd(sbuf + (size_t)dN * HH + h, ex);
}

// ---------------------------------------------------------------------------
// Pass C: one wave per edge. msg = V[src] * attn, scatter-add into hout[dst].
// Lane l handles 8 consecutive floats (head = l>>2).
// ---------------------------------------------------------------------------
__global__ __launch_bounds__(256) void edge_scatter_kernel(
    const float* __restrict__ V, const float* __restrict__ scores,
    const float* __restrict__ sbuf, const int* __restrict__ src,
    const int* __restrict__ dst, float* __restrict__ hout) {
  int wid = blockIdx.x * 8 + (threadIdx.x >> 5);
  if (wid >= EE) return;
  int l  = threadIdx.x & 31;
  int sN = src[wid], dN = dst[wid];
  int h  = l >> 2;
  float attn = scores[(size_t)wid * HH + h] / sbuf[(size_t)dN * HH + h];
  const float4* vr = (const float4*)(V + (size_t)sN * DD);
  float4 v0 = vr[2 * l], v1 = vr[2 * l + 1];
  float* op = hout + (size_t)dN * DD + l * 8;
  atomicAdd(op + 0, v0.x * attn);
  atomicAdd(op + 1, v0.y * attn);
  atomicAdd(op + 2, v0.z * attn);
  atomicAdd(op + 3, v0.w * attn);
  atomicAdd(op + 4, v1.x * attn);
  atomicAdd(op + 5, v1.y * attn);
  atomicAdd(op + 6, v1.z * attn);
  atomicAdd(op + 7, v1.w * attn);
}

// ---------------------------------------------------------------------------
// Output projection + sigmoid-skip blend + LayerNorm, fully fused.
// Block owns a complete 16x256 tile -> row stats via shfl + LDS reductions.
// ---------------------------------------------------------------------------
__global__ __launch_bounds__(256) void wmma_gemm_ln_kernel(
    const float* __restrict__ A, const __bf16* __restrict__ Bp,
    const float* __restrict__ bias, const float* __restrict__ resid,
    const float* __restrict__ skip, const float* __restrict__ lnw,
    const float* __restrict__ lnb, float* __restrict__ out) {
  __shared__ float red_s[8][16];
  __shared__ float red_q[8][16];
  __shared__ float s_mean[16];
  __shared__ float s_rstd[16];
  const int w    = threadIdx.x >> 5;
  const int l    = threadIdx.x & 31;
  const int lrow = l & 15;
  const int hi   = l >> 4;
  const int m0   = blockIdx.x * 16;
  v8f c0 = {0, 0, 0, 0, 0, 0, 0, 0};
  v8f c1 = {0, 0, 0, 0, 0, 0, 0, 0};
  const float*  arow = A + (size_t)(m0 + lrow) * DD + hi * 8;
  const __bf16* bp0  = Bp + ((size_t)(2 * w) * 8 * 32 + l) * 16;
  const __bf16* bp1  = Bp + ((size_t)(2 * w + 1) * 8 * 32 + l) * 16;
#pragma unroll
  for (int kk = 0; kk < 8; ++kk) {
    v16bf af = load_afrag(arow + kk * 32);
    v16bf b0 = *(const v16bf*)(bp0 + (size_t)kk * 32 * 16);
    v16bf b1 = *(const v16bf*)(bp1 + (size_t)kk * 32 * 16);
    c0 = __builtin_amdgcn_wmma_f32_16x16x32_bf16(false, af, false, b0, (short)0, c0, false, false);
    c1 = __builtin_amdgcn_wmma_f32_16x16x32_bf16(false, af, false, b1, (short)0, c1, false, false);
  }
  const int   n0 = w * 32 + lrow, n1 = n0 + 16;
  const float bb0 = bias[n0], bb1 = bias[n1];
  const float alpha = 1.0f / (1.0f + __expf(-skip[0]));
  const float beta  = 1.0f - alpha;
  const int   rbase = m0 + hi * 8;
  float v0a[8], v1a[8];
#pragma unroll
  for (int r = 0; r < 8; ++r) {
    const float* rr = resid + (size_t)(rbase + r) * DD;
    v0a[r] = alpha * (c0[r] + bb0) + beta * rr[n0];
    v1a[r] = alpha * (c1[r] + bb1) + beta * rr[n1];
  }
#pragma unroll
  for (int r = 0; r < 8; ++r) {
    float sr = v0a[r] + v1a[r];
    float qr = v0a[r] * v0a[r] + v1a[r] * v1a[r];
#pragma unroll
    for (int m = 1; m < 16; m <<= 1) {
      sr += __shfl_xor(sr, m, 32);
      qr += __shfl_xor(qr, m, 32);
    }
    if (lrow == 0) { red_s[w][hi * 8 + r] = sr; red_q[w][hi * 8 + r] = qr; }
  }
  __syncthreads();
  if (threadIdx.x < 16) {
    float ts = 0.f, tq = 0.f;
#pragma unroll
    for (int ww = 0; ww < 8; ++ww) { ts += red_s[ww][threadIdx.x]; tq += red_q[ww][threadIdx.x]; }
    float mean = ts * (1.0f / DD);
    float var  = tq * (1.0f / DD) - mean * mean;
    s_mean[threadIdx.x] = mean;
    s_rstd[threadIdx.x] = rsqrtf(var + 1e-5f);
  }
  __syncthreads();
  const float lw0 = lnw[n0], lb0 = lnb[n0], lw1 = lnw[n1], lb1 = lnb[n1];
#pragma unroll
  for (int r = 0; r < 8; ++r) {
    int   row = hi * 8 + r;
    float mn = s_mean[row], rs = s_rstd[row];
    out[(size_t)(rbase + r) * DD + n0] = (v0a[r] - mn) * rs * lw0 + lb0;
    out[(size_t)(rbase + r) * DD + n1] = (v1a[r] - mn) * rs * lw1 + lb1;
  }
}

// ---------------------------------------------------------------------------
extern "C" void kernel_launch(void* const* d_in, const int* in_sizes, int n_in,
                              void* d_out, int out_size, void* d_ws, size_t ws_size,
                              hipStream_t stream) {
  (void)in_sizes; (void)n_in; (void)out_size; (void)ws_size;
  const float* feats = (const float*)d_in[0];
  const float* Wk    = (const float*)d_in[1];
  const float* bk    = (const float*)d_in[2];
  const float* Wq    = (const float*)d_in[3];
  const float* bq    = (const float*)d_in[4];
  const float* Wv    = (const float*)d_in[5];
  const float* bv    = (const float*)d_in[6];
  const float* Wa    = (const float*)d_in[7];
  const float* ba    = (const float*)d_in[8];
  const float* lnw   = (const float*)d_in[9];
  const float* lnb   = (const float*)d_in[10];
  const float* skip  = (const float*)d_in[11];
  const float* mu    = (const float*)d_in[12];
  const float* watt  = (const float*)d_in[13];
  const float* wmsg  = (const float*)d_in[14];
  const int*   src   = (const int*)d_in[15];
  const int*   dst   = (const int*)d_in[16];

  char*  ws  = (char*)d_ws;
  size_t off = 0;
  auto take = [&](size_t bytes) -> char* {
    char* p = ws + off;
    off += (bytes + 255) & ~(size_t)255;
    return p;
  };
  float*  WkF  = (float*)take((size_t)2 * DD * DD * sizeof(float));
  float*  WvF  = (float*)take((size_t)2 * DD * DD * sizeof(float));
  float*  bkF  = (float*)take((size_t)2 * DD * sizeof(float));
  float*  bvF  = (float*)take((size_t)2 * DD * sizeof(float));
  __bf16* BpK[2]; __bf16* BpV[2]; __bf16* BpQ[2]; __bf16* BpA[2];
  for (int i = 0; i < 2; ++i) BpK[i] = (__bf16*)take((size_t)DD * DD * sizeof(__bf16));
  for (int i = 0; i < 2; ++i) BpV[i] = (__bf16*)take((size_t)DD * DD * sizeof(__bf16));
  for (int i = 0; i < 2; ++i) BpQ[i] = (__bf16*)take((size_t)DD * DD * sizeof(__bf16));
  for (int i = 0; i < 2; ++i) BpA[i] = (__bf16*)take((size_t)DD * DD * sizeof(__bf16));
  float* Kbuf   = (float*)take((size_t)NN * DD * sizeof(float));
  float* Vbuf   = (float*)take((size_t)NN * DD * sizeof(float));
  float* Qbuf   = (float*)take((size_t)NN * DD * sizeof(float));
  float* hout   = (float*)take((size_t)2 * NN * DD * sizeof(float));
  float* scores = (float*)take((size_t)EE * HH * sizeof(float));
  float* mbuf   = (float*)take((size_t)NN * HH * sizeof(float));
  float* sbuf   = (float*)take((size_t)NN * HH * sizeof(float));

  // 1. fuse per-head w_att/w_msg into projection weights, pack all B matrices
  fuse_weights_kernel<<<(2 * DD * DD) / 256, 256, 0, stream>>>(
      Wk, bk, Wv, bv, watt, wmsg, WkF, bkF, WvF, bvF);
  for (int et = 0; et < 2; ++et) {
    pack_b_kernel<<<128, 32, 0, stream>>>(WkF + (size_t)et * DD * DD, BpK[et]);
    pack_b_kernel<<<128, 32, 0, stream>>>(WvF + (size_t)et * DD * DD, BpV[et]);
    pack_b_kernel<<<128, 32, 0, stream>>>(Wq + (size_t)et * DD * DD, BpQ[et]);
    pack_b_kernel<<<128, 32, 0, stream>>>(Wa + (size_t)et * DD * DD, BpA[et]);
  }

  const float mscale = 0.17677669529663687f;  // 1/sqrt(32)

  // 2. per-edge-type: projections (WMMA) then softmax-aggregation over edges
  for (int et = 0; et < 2; ++et) {
    int dt = 1 - et;
    wmma_gemm_bias_kernel<<<NN / 16, 256, 0, stream>>>(
        feats + (size_t)et * NN * DD, BpK[et], bkF + et * DD, Kbuf);
    wmma_gemm_bias_kernel<<<NN / 16, 256, 0, stream>>>(
        feats + (size_t)et * NN * DD, BpV[et], bvF + et * DD, Vbuf);
    wmma_gemm_bias_kernel<<<NN / 16, 256, 0, stream>>>(
        feats + (size_t)dt * NN * DD, BpQ[dt], bq + dt * DD, Qbuf);
    init_et_kernel<<<(NN * DD) / 256, 256, 0, stream>>>(
        mbuf, sbuf, hout + (size_t)dt * NN * DD);
    edge_scores_kernel<<<EE / 8, 256, 0, stream>>>(
        Qbuf, Kbuf, src + (size_t)et * EE, dst + (size_t)et * EE,
        mu + et * HH, mscale, scores, mbuf);
    edge_exp_kernel<<<(EE * HH) / 256, 256, 0, stream>>>(
        scores, dst + (size_t)et * EE, mbuf, sbuf);
    edge_scatter_kernel<<<EE / 8, 256, 0, stream>>>(
        Vbuf, scores, sbuf, src + (size_t)et * EE, dst + (size_t)et * EE,
        hout + (size_t)dt * NN * DD);
  }

  // 3. fused output projection + skip blend + LayerNorm -> d_out
  for (int nt = 0; nt < 2; ++nt) {
    wmma_gemm_ln_kernel<<<NN / 16, 256, 0, stream>>>(
        hout + (size_t)nt * NN * DD, BpA[nt], ba + nt * DD,
        feats + (size_t)nt * NN * DD, skip + nt,
        lnw + nt * DD, lnb + nt * DD,
        (float*)d_out + (size_t)nt * NN * DD);
  }
}
